// Attention_3848290697244
// MI455X (gfx1250) — matrix operand order repaired and checked by reading, not compiled, not run on hardware
//
#include <hip/hip_runtime.h>
#include <hip/hip_bf16.h>
#include <float.h>

// MI455X (gfx1250) fused attention:
//   cvt(f32->bf16)  ->  qkv_gemm (WMMA)  ->  flash attn (WMMA + LDS transpose)
//   -> out_gemm (WMMA, fp32 out)
// All matmuls use v_wmma_f32_16x16x32_bf16 (f32 accumulate).

typedef __attribute__((ext_vector_type(16))) __bf16 bf16x16;
typedef __attribute__((ext_vector_type(8)))  __bf16 bf16x8;
typedef __attribute__((ext_vector_type(8)))  float  f32x8;

#define DEV __device__ __forceinline__

DEV f32x8 wmma_bf16(bf16x16 a, bf16x16 b, f32x8 c) {
  return __builtin_amdgcn_wmma_f32_16x16x32_bf16(
      /*neg_a=*/false, a, /*neg_b=*/false, b,
      /*c_mod=*/(short)0, c, /*reuse_a=*/false, /*reuse_b=*/false);
}

// A-matrix fragment (16x32 bf16, row-major source, leading dim `ld`):
// lane r (0..15) holds row `row0+r`; half h: K in {0..7,16..23} + 8h  (ISA 7.12.2)
DEV bf16x16 load_fragA(const __bf16* p, int ld, int row0, int k0) {
  const int lane = threadIdx.x & 31;
  const int r = lane & 15, h = lane >> 4;
  const __bf16* rp = p + (size_t)(row0 + r) * ld + k0 + h * 8;
  bf16x8 lo = *reinterpret_cast<const bf16x8*>(rp);
  bf16x8 hi = *reinterpret_cast<const bf16x8*>(rp + 16);
  bf16x16 f;
#pragma unroll
  for (int i = 0; i < 8; ++i) { f[i] = lo[i]; f[i + 8] = hi[i]; }
  return f;
}

// B-matrix fragment (32x16 bf16) built as B = W^T from row-major W [N x K]:
// lane r holds output column `col0+r`; half h: K in {0..7,16..23} + 8h (matches A)
DEV bf16x16 load_fragB(const __bf16* p, int ld, int col0, int k0) {
  const int lane = threadIdx.x & 31;
  const int r = lane & 15, h = lane >> 4;
  const __bf16* rp = p + (size_t)(col0 + r) * ld + k0 + h * 8;
  bf16x8 lo = *reinterpret_cast<const bf16x8*>(rp);
  bf16x8 hi = *reinterpret_cast<const bf16x8*>(rp + 16);
  bf16x16 f;
#pragma unroll
  for (int i = 0; i < 8; ++i) { f[i] = lo[i]; f[i + 8] = hi[i]; }
  return f;
}

// ---------------------------------------------------------------- convert
__global__ __launch_bounds__(256) void cvt_f32_bf16(const float* __restrict__ in,
                                                    __bf16* __restrict__ out, int n) {
  int i = blockIdx.x * 256 + threadIdx.x;
  if (i < n) out[i] = (__bf16)in[i];
}

// ---------------------------------------------------------------- QKV GEMM
// C[4096,3072] = Xb[4096,1024] * Wqkv^T ; scatter into q/k (row-major [n,64])
// and v transposed [64,n] per (b,head).
__global__ __launch_bounds__(256) void qkv_gemm(const __bf16* __restrict__ xb,
                                                const __bf16* __restrict__ wq,
                                                __bf16* __restrict__ qb,
                                                __bf16* __restrict__ kb,
                                                __bf16* __restrict__ vt) {
  const int wave = threadIdx.x >> 5;
  const int m0 = blockIdx.y * 128 + (wave & 3) * 32;
  const int n0 = blockIdx.x * 128 + (wave >> 2) * 64;

  f32x8 acc[2][4] = {};
  for (int k = 0; k < 1024; k += 32) {
    bf16x16 a0 = load_fragA(xb, 1024, m0, k);
    bf16x16 a1 = load_fragA(xb, 1024, m0 + 16, k);
#pragma unroll
    for (int j = 0; j < 4; ++j) {
      bf16x16 b = load_fragB(wq, 1024, n0 + 16 * j, k);
      acc[0][j] = wmma_bf16(a0, b, acc[0][j]);
      acc[1][j] = wmma_bf16(a1, b, acc[1][j]);
    }
  }

  const int lane = threadIdx.x & 31;
  const int r = lane & 15, h = lane >> 4;
#pragma unroll
  for (int i = 0; i < 2; ++i)
#pragma unroll
    for (int j = 0; j < 4; ++j)
#pragma unroll
      for (int v = 0; v < 8; ++v) {
        int row = m0 + 16 * i + v + 8 * h;   // b*2048 + n
        int col = n0 + 16 * j + r;           // e in [0,3072)
        int bidx = row >> 11, n = row & 2047;
        int which = col >> 10, rem = col & 1023;
        int head = rem >> 6, d = rem & 63;
        size_t hb = (size_t)(bidx * 16 + head);
        __bf16 val = (__bf16)acc[i][j][v];
        if (which == 0)      qb[(hb * 2048 + n) * 64 + d] = val;
        else if (which == 1) kb[(hb * 2048 + n) * 64 + d] = val;
        else                 vt[(hb * 64 + d) * 2048 + n] = val;
      }
}

// ---------------------------------------------------------------- attention
// Per wave: 16 query rows, online softmax over key tiles of 64.
__global__ __launch_bounds__(256) void attn_kernel(const __bf16* __restrict__ qb,
                                                   const __bf16* __restrict__ kb,
                                                   const __bf16* __restrict__ vt,
                                                   const int* __restrict__ mask,
                                                   __bf16* __restrict__ ob) {
  __shared__ __bf16 pLds[8 * 16 * 64];  // per-wave 16x64 P staging (C->A relayout)
  const float kScale = 0.125f;          // 64^-0.5
  const int wave = threadIdx.x >> 5;
  const int lane = threadIdx.x & 31;
  const int r = lane & 15, hh = lane >> 4;
  const int bh = blockIdx.y;            // b*16 + head
  const int b = bh >> 4, head = bh & 15;
  const int q0 = blockIdx.x * 128 + wave * 16;

  const __bf16* qh = qb + (size_t)bh * 2048 * 64;
  const __bf16* kh = kb + (size_t)bh * 2048 * 64;
  const __bf16* vh = vt + (size_t)bh * 64 * 2048;
  const int* mkp = mask + b * 2048;

  bf16x16 qa0 = load_fragA(qh, 64, q0, 0);
  bf16x16 qa1 = load_fragA(qh, 64, q0, 32);

  int mq[8];
#pragma unroll
  for (int v = 0; v < 8; ++v) mq[v] = mkp[q0 + v + 8 * hh];

  float m[8], l[8];
  f32x8 oacc[4] = {};
#pragma unroll
  for (int v = 0; v < 8; ++v) { m[v] = -FLT_MAX; l[v] = 0.f; }

  __bf16* pw = pLds + wave * 16 * 64;

  for (int j0 = 0; j0 < 2048; j0 += 64) {
    // S = Q * K^T  (16 x 64 tile of scores)
    f32x8 s[4] = {};
#pragma unroll
    for (int nt = 0; nt < 4; ++nt) {
      bf16x16 b0 = load_fragB(kh, 64, j0 + nt * 16, 0);
      s[nt] = wmma_bf16(qa0, b0, s[nt]);
      bf16x16 b1 = load_fragB(kh, 64, j0 + nt * 16, 32);
      s[nt] = wmma_bf16(qa1, b1, s[nt]);
    }
    int mk[4];
#pragma unroll
    for (int nt = 0; nt < 4; ++nt) mk[nt] = mkp[j0 + nt * 16 + r];

    // online softmax (row = v+8*hh lives across lanes of one half)
    float corr[8];
#pragma unroll
    for (int v = 0; v < 8; ++v) {
      float tm = -FLT_MAX;
#pragma unroll
      for (int nt = 0; nt < 4; ++nt) {
        float sv = s[nt][v] * kScale;
        sv = (mq[v] != 0 && mk[nt] != 0) ? sv : -FLT_MAX;
        s[nt][v] = sv;
        tm = fmaxf(tm, sv);
      }
      tm = fmaxf(tm, __shfl_xor(tm, 1));
      tm = fmaxf(tm, __shfl_xor(tm, 2));
      tm = fmaxf(tm, __shfl_xor(tm, 4));
      tm = fmaxf(tm, __shfl_xor(tm, 8));
      float mn = fmaxf(m[v], tm);
      corr[v] = __expf(m[v] - mn);
      float rs = 0.f;
#pragma unroll
      for (int nt = 0; nt < 4; ++nt) {
        float p = __expf(s[nt][v] - mn);
        s[nt][v] = p;
        rs += p;
      }
      rs += __shfl_xor(rs, 1);
      rs += __shfl_xor(rs, 2);
      rs += __shfl_xor(rs, 4);
      rs += __shfl_xor(rs, 8);
      l[v] = l[v] * corr[v] + rs;
      m[v] = mn;
    }
#pragma unroll
    for (int dt = 0; dt < 4; ++dt)
#pragma unroll
      for (int v = 0; v < 8; ++v) oacc[dt][v] *= corr[v];

    // P: C-layout -> LDS -> A-fragment layout (same-wave LDS is in-order)
#pragma unroll
    for (int v = 0; v < 8; ++v)
#pragma unroll
      for (int nt = 0; nt < 4; ++nt)
        pw[(v + 8 * hh) * 64 + nt * 16 + r] = (__bf16)s[nt][v];
    asm volatile("s_wait_dscnt 0" ::: "memory");
    bf16x16 pa0 = load_fragA(pw, 64, 0, 0);
    bf16x16 pa1 = load_fragA(pw, 64, 0, 32);

    // O += P * V   (V stored transposed [64, n] -> contiguous B loads)
#pragma unroll
    for (int dt = 0; dt < 4; ++dt) {
      bf16x16 vb0 = load_fragB(vh, 2048, dt * 16, j0);
      oacc[dt] = wmma_bf16(pa0, vb0, oacc[dt]);
      bf16x16 vb1 = load_fragB(vh, 2048, dt * 16, j0 + 32);
      oacc[dt] = wmma_bf16(pa1, vb1, oacc[dt]);
    }
  }

  // normalize and store to o[b*2048+n, head*64+d] (bf16)
#pragma unroll
  for (int dt = 0; dt < 4; ++dt)
#pragma unroll
    for (int v = 0; v < 8; ++v) {
      float o = oacc[dt][v] / l[v];
      size_t row = (size_t)(b * 2048 + q0 + v + 8 * hh);
      ob[row * 1024 + head * 64 + dt * 16 + r] = (__bf16)o;
    }
}

// ---------------------------------------------------------------- out proj
// out[4096,1024] (fp32) = Ob[4096,1024] * Wout^T
__global__ __launch_bounds__(256) void out_gemm(const __bf16* __restrict__ ob,
                                                const __bf16* __restrict__ wo,
                                                float* __restrict__ out) {
  const int wave = threadIdx.x >> 5;
  const int m0 = blockIdx.y * 128 + (wave & 3) * 32;
  const int n0 = blockIdx.x * 128 + (wave >> 2) * 64;

  f32x8 acc[2][4] = {};
  for (int k = 0; k < 1024; k += 32) {
    bf16x16 a0 = load_fragA(ob, 1024, m0, k);
    bf16x16 a1 = load_fragA(ob, 1024, m0 + 16, k);
#pragma unroll
    for (int j = 0; j < 4; ++j) {
      bf16x16 b = load_fragB(wo, 1024, n0 + 16 * j, k);
      acc[0][j] = wmma_bf16(a0, b, acc[0][j]);
      acc[1][j] = wmma_bf16(a1, b, acc[1][j]);
    }
  }

  const int lane = threadIdx.x & 31;
  const int r = lane & 15, h = lane >> 4;
#pragma unroll
  for (int i = 0; i < 2; ++i)
#pragma unroll
    for (int j = 0; j < 4; ++j)
#pragma unroll
      for (int v = 0; v < 8; ++v) {
        size_t row = (size_t)(m0 + 16 * i + v + 8 * h);
        out[row * 1024 + n0 + 16 * j + r] = acc[i][j][v];
      }
}

// ---------------------------------------------------------------- launch
extern "C" void kernel_launch(void* const* d_in, const int* in_sizes, int n_in,
                              void* d_out, int out_size, void* d_ws, size_t ws_size,
                              hipStream_t stream) {
  const float* x     = (const float*)d_in[0];  // [2,2048,1024]
  const int*   mask  = (const int*)d_in[1];    // [2,2048]
  const float* w_qkv = (const float*)d_in[2];  // [3072,1024]
  const float* w_out = (const float*)d_in[3];  // [1024,1024]
  float* out = (float*)d_out;                  // [2,2048,1024]

  // workspace layout (bytes), ~36 MB total
  char* ws = (char*)d_ws;
  __bf16* xb    = (__bf16*)(ws + 0);          // 4096*1024     = 8 MB
  __bf16* wqkvb = (__bf16*)(ws + 8388608);    // 3072*1024     = 6 MB
  __bf16* woutb = (__bf16*)(ws + 14680064);   // 1024*1024     = 2 MB
  __bf16* qb    = (__bf16*)(ws + 16777216);   // [bh,2048,64]  = 8 MB
  __bf16* kb    = (__bf16*)(ws + 20971520);   // [bh,2048,64]  = 8 MB
  __bf16* vt    = (__bf16*)(ws + 25165824);   // [bh,64,2048]  = 8 MB (transposed)
  __bf16* obuf  = (__bf16*)(ws + 29360128);   // [4096,1024]   = 8 MB

  cvt_f32_bf16<<<(4194304 + 255) / 256, 256, 0, stream>>>(x, xb, 4194304);
  cvt_f32_bf16<<<(3145728 + 255) / 256, 256, 0, stream>>>(w_qkv, wqkvb, 3145728);
  cvt_f32_bf16<<<(1048576 + 255) / 256, 256, 0, stream>>>(w_out, woutb, 1048576);

  qkv_gemm<<<dim3(24, 32), 256, 0, stream>>>(xb, wqkvb, qb, kb, vt);
  attn_kernel<<<dim3(16, 32), 256, 0, stream>>>(qb, kb, vt, mask, obuf);
  out_gemm<<<dim3(8, 32), 256, 0, stream>>>(obuf, woutb, out);
}
